// Encoder3DFactorizedSelfAttentionBlock_13494787244157
// MI455X (gfx1250) — matrix operand order repaired and checked by reading, not compiled, not run on hardware
//
#include <hip/hip_runtime.h>
#include <hip/hip_bf16.h>
#include <cstdint>

// ---------------------------------------------------------------------------
// Encoder3D factorized self-attention block for MI455X (gfx1250, wave32).
// Residual stream fp32 in d_out; all GEMMs via v_wmma_f32_16x16x32_f16 with
// LDS tiling fed by GLOBAL_LOAD_ASYNC_TO_LDS_B128 (double-buffered, ASYNCcnt).
// ---------------------------------------------------------------------------

typedef _Float16 h16;
typedef __attribute__((ext_vector_type(16))) _Float16 v16h;
typedef __attribute__((ext_vector_type(8)))  float    v8f;
typedef __attribute__((ext_vector_type(4)))  unsigned int u32x4;

union FragAB { v16h h; u32x4 q[2]; };

static __device__ __forceinline__ v16h load_frag(const h16* lo, const h16* hi) {
  FragAB f;
  f.q[0] = *(const u32x4*)lo;
  f.q[1] = *(const u32x4*)hi;
  return f.h;
}

static __device__ __forceinline__ v8f wmma_f16(v16h a, v16h b, v8f c) {
  return __builtin_amdgcn_wmma_f32_16x16x32_f16(false, a, false, b, (short)0, c,
                                                false, false);
}

// Issue one 16-byte global -> LDS async DMA (tracked with ASYNCcnt).
static __device__ __forceinline__ void async_b128(const h16* gsrc, const h16* ldst) {
  unsigned lds_addr = (unsigned)(uintptr_t)ldst;  // low 32 bits = LDS byte addr
  asm volatile("global_load_async_to_lds_b128 %0, %1, off"
               :: "v"(lds_addr), "v"(gsrc) : "memory");
}
static __device__ __forceinline__ void wait_async0() {
  asm volatile("s_wait_asynccnt 0" ::: "memory");
}

// ---------------------------------------------------------------------------
// Weight transpose + f32->f16 convert: src [K,N] row-major -> dst [N,K]
// ---------------------------------------------------------------------------
__global__ void tconv_kernel(const float* __restrict__ src, h16* __restrict__ dst,
                             int K, int N) {
  int i = blockIdx.x * 256 + threadIdx.x;
  if (i >= K * N) return;
  int n = i / K, k = i % K;
  dst[i] = (h16)src[(size_t)k * N + n];
}

// ---------------------------------------------------------------------------
// LayerNorm over C=512, fp32 in -> f16 out. One wave per token (32 lanes x 16).
// ---------------------------------------------------------------------------
__global__ __launch_bounds__(256)
void ln_to_f16(const float* __restrict__ x, const float* __restrict__ sc,
               const float* __restrict__ bi, h16* __restrict__ out, int ntok) {
  const int lane = threadIdx.x & 31;
  const int wave = threadIdx.x >> 5;
  const int tok  = blockIdx.x * 8 + wave;
  if (tok >= ntok) return;
  const float* p = x + (size_t)tok * 512 + lane * 16;
  float4 r[4];
#pragma unroll
  for (int i = 0; i < 4; ++i) r[i] = ((const float4*)p)[i];
  float s = 0.f, ss = 0.f;
#pragma unroll
  for (int i = 0; i < 4; ++i) {
    s  += r[i].x + r[i].y + r[i].z + r[i].w;
    ss += r[i].x*r[i].x + r[i].y*r[i].y + r[i].z*r[i].z + r[i].w*r[i].w;
  }
#pragma unroll
  for (int o = 16; o > 0; o >>= 1) {
    s  += __shfl_xor(s,  o, 32);
    ss += __shfl_xor(ss, o, 32);
  }
  const float mean = s * (1.f / 512.f);
  const float var  = ss * (1.f / 512.f) - mean * mean;
  const float rstd = rsqrtf(var + 1e-6f);
  const float* scp = sc + lane * 16;
  const float* bip = bi + lane * 16;
  v16h o16;
#pragma unroll
  for (int i = 0; i < 16; ++i) {
    float xv = ((const float*)r)[i];
    o16[i] = (h16)((xv - mean) * rstd * scp[i] + bip[i]);
  }
  *(v16h*)(out + (size_t)tok * 512 + lane * 16) = o16;
}

// ---------------------------------------------------------------------------
// LDS-tiled WMMA GEMM: Y[M,N] = A[M,K](f16) @ BT[N,K](f16)^T + bias.
// Block: 256 thr (8 waves), tile 128x64, BK=32, double-buffered async DMA.
// Wave grid 4m x 2n; each wave owns a 32x32 patch (4 accumulators).
// Epilogues: 0 = f16*(alpha), 1 = fp32 residual +=, 2 = gelu_tanh -> f16.
// ---------------------------------------------------------------------------
template<int EPI>
__global__ __launch_bounds__(256)
void gemm_wmma_tiled(const h16* __restrict__ A, const h16* __restrict__ BT,
                     const float* __restrict__ bias,
                     h16* __restrict__ outh, float* __restrict__ resid,
                     int M, int N, int K, float alpha) {
  __shared__ h16 Al[2][128][40];   // padded rows (80B stride)
  __shared__ h16 Bl[2][64][40];

  const int tid  = threadIdx.x;
  const int lane = tid & 31;
  const int wave = tid >> 5;
  const int wm   = wave & 3;
  const int wn   = wave >> 2;
  const int bm   = blockIdx.y * 128;
  const int bnn  = blockIdx.x * 64;

  // staging chunks: A = 512 x 16B chunks (2/thread), B = 256 x 16B (1/thread)
  const int a_r0 = tid >> 2,          a_c0 = (tid & 3) * 8;
  const int a_r1 = (tid + 256) >> 2,  a_c1 = ((tid + 256) & 3) * 8;
  const int b_r  = tid >> 2,          b_c  = (tid & 3) * 8;

  const h16* gA0 = A  + (size_t)(bm  + a_r0) * K + a_c0;
  const h16* gA1 = A  + (size_t)(bm  + a_r1) * K + a_c1;
  const h16* gB  = BT + (size_t)(bnn + b_r ) * K + b_c;

  const int ka = (lane >= 16) ? 8 : 0;    // A frag K split (ISA A layout)
  const int kb = (lane >= 16) ? 8 : 0;    // B frag K split (ISA B layout)
  const int arow0 = wm * 32 + (lane & 15);
  const int arow1 = arow0 + 16;
  const int bcol0 = wn * 32 + (lane & 15);
  const int bcol1 = bcol0 + 16;

  v8f acc00 = {}, acc01 = {}, acc10 = {}, acc11 = {};

  // prologue: DMA chunk 0 into buffer 0
  async_b128(gA0, &Al[0][a_r0][a_c0]);
  async_b128(gA1, &Al[0][a_r1][a_c1]);
  async_b128(gB,  &Bl[0][b_r ][b_c ]);

  const int KC = K >> 5;
  for (int kc = 0; kc < KC; ++kc) {
    const int cur = kc & 1;
    wait_async0();        // my DMA for chunk kc has landed in LDS
    __syncthreads();      // everyone's DMA landed; everyone done reading buf^1
    if (kc + 1 < KC) {    // overlap: DMA chunk kc+1 while computing chunk kc
      const int k1 = (kc + 1) << 5;
      async_b128(gA0 + k1, &Al[cur ^ 1][a_r0][a_c0]);
      async_b128(gA1 + k1, &Al[cur ^ 1][a_r1][a_c1]);
      async_b128(gB  + k1, &Bl[cur ^ 1][b_r ][b_c ]);
    }
    v16h a0 = load_frag(&Al[cur][arow0][ka], &Al[cur][arow0][ka + 16]);
    v16h a1 = load_frag(&Al[cur][arow1][ka], &Al[cur][arow1][ka + 16]);
    v16h b0 = load_frag(&Bl[cur][bcol0][kb], &Bl[cur][bcol0][kb + 16]);
    v16h b1 = load_frag(&Bl[cur][bcol1][kb], &Bl[cur][bcol1][kb + 16]);
    acc00 = wmma_f16(a0, b0, acc00);
    acc01 = wmma_f16(a0, b1, acc01);
    acc10 = wmma_f16(a1, b0, acc10);
    acc11 = wmma_f16(a1, b1, acc11);
  }

  const int n0  = bnn + wn * 32 + (lane & 15);
  const int n1  = n0 + 16;
  const float bv0 = bias[n0];
  const float bv1 = bias[n1];
  const int mr = bm + wm * 32 + ((lane >= 16) ? 8 : 0);

#pragma unroll
  for (int i = 0; i < 2; ++i) {
    const v8f* accs[2][2] = {{&acc00, &acc01}, {&acc10, &acc11}};
#pragma unroll
    for (int j = 0; j < 2; ++j) {
      const v8f& acc = *accs[i][j];
      const int   nn = (j == 0) ? n0 : n1;
      const float bb = (j == 0) ? bv0 : bv1;
#pragma unroll
      for (int vv = 0; vv < 8; ++vv) {
        float  val = acc[vv] + bb;
        size_t idx = (size_t)(mr + i * 16 + vv) * N + nn;
        if (EPI == 0) {
          outh[idx] = (h16)(val * alpha);
        } else if (EPI == 1) {
          resid[idx] += val;
        } else {
          float t = tanhf(0.7978845608028654f * (val + 0.044715f * val * val * val));
          outh[idx] = (h16)(0.5f * val * (1.0f + t));
        }
      }
    }
  }
}

// ---------------------------------------------------------------------------
// Axial attention over one (sequence, head). S = 16 (time) or 32 (H/W).
// token(seq,pos) = (seq>>shift)*mul + (seq & ((1<<shift)-1)) + pos*stride
// ---------------------------------------------------------------------------
template<int S>
__global__ __launch_bounds__(256)
void axial_attn(const h16* __restrict__ q, const h16* __restrict__ k,
                const h16* __restrict__ v, h16* __restrict__ o,
                int stride, int shift, int mul) {
  __shared__ h16   kl[32][64];       // K rows (feature-contiguous)  4 KB
  __shared__ h16   vT[64][32];       // V^T (key-pos contiguous)     4 KB
  __shared__ float sS[8][16][32];    // per-wave score staging      16 KB
  __shared__ h16   pP[8][16][32];    // per-wave softmax(P) f16      8 KB

  const int head = blockIdx.x;
  const int seq  = blockIdx.y;
  const int base = (seq >> shift) * mul + (seq & ((1 << shift) - 1));
  const int tid  = threadIdx.x;
  const int lane = tid & 31;
  const int wave = tid >> 5;
  const int nthr = blockDim.x;
  constexpr int MT = S / 16;
  const int mt    = wave % MT;
  const int ntile = wave / MT;
  const int hoff  = head * 64;

  // zero pads (LDS is uninitialized; 0*NaN would poison the AV WMMA)
  for (int i = tid; i < 32 * 64; i += nthr) ((h16*)kl)[i] = (h16)0.f;
  for (int i = tid; i < 64 * 32; i += nthr) ((h16*)vT)[i] = (h16)0.f;
  __syncthreads();
  for (int i = tid; i < S * 64; i += nthr) {
    int s = i >> 6, d = i & 63;
    size_t g = (size_t)(base + s * stride) * 512 + hoff + d;
    kl[s][d] = k[g];
    vT[d][s] = v[g];
  }
  __syncthreads();

  const int ka = (lane >= 16) ? 8 : 0;
  const int kb = (lane >= 16) ? 8 : 0;
  const int rowq = base + (mt * 16 + (lane & 15)) * stride;
  const h16* pq = q + (size_t)rowq * 512 + hoff;

  // scores: (16 x S) = q-tile(16x64) @ k^T(64xS), K=64 -> 2 WMMAs per n-half
#pragma unroll
  for (int nh = 0; nh < S / 16; ++nh) {
    v8f sacc = {};
#pragma unroll
    for (int kc = 0; kc < 2; ++kc) {
      v16h a = load_frag(pq + kc * 32 + ka, pq + kc * 32 + ka + 16);
      const h16* pk = &kl[nh * 16 + (lane & 15)][kc * 32 + kb];
      v16h b = load_frag(pk, pk + 16);
      sacc = wmma_f16(a, b, sacc);
    }
    const int ncol = nh * 16 + (lane & 15);
    const int mr0  = (lane >= 16) ? 8 : 0;
#pragma unroll
    for (int vv = 0; vv < 8; ++vv) sS[wave][mr0 + vv][ncol] = sacc[vv];
  }
  __syncthreads();

  // row softmax (lanes 0..15 own one row each of this wave's m-tile)
  if (lane < 16) {
    const int r = lane;
    float mx = -1e30f;
#pragma unroll
    for (int c = 0; c < S; ++c) mx = fmaxf(mx, sS[wave][r][c]);
    float sum = 0.f;
#pragma unroll
    for (int c = 0; c < S; ++c) sum += __expf(sS[wave][r][c] - mx);
    const float inv = 1.f / sum;
#pragma unroll
    for (int c = 0; c < S; ++c) pP[wave][r][c] = (h16)(__expf(sS[wave][r][c] - mx) * inv);
#pragma unroll
    for (int c = S; c < 32; ++c) pP[wave][r][c] = (h16)0.f;
  }
  __syncthreads();

  // AV: out(16x16) = P(16xS, zero-padded to K=32) @ V(Sx64) n-tile
  const h16* pp = &pP[wave][lane & 15][ka];
  v16h aP = load_frag(pp, pp + 16);
  const h16* pv = &vT[ntile * 16 + (lane & 15)][kb];
  v16h bV = load_frag(pv, pv + 16);
  v8f oacc = {};
  oacc = wmma_f16(aP, bV, oacc);

  const int col = hoff + ntile * 16 + (lane & 15);
  const int mr0 = mt * 16 + ((lane >= 16) ? 8 : 0);
#pragma unroll
  for (int vv = 0; vv < 8; ++vv) {
    size_t g = (size_t)(base + (mr0 + vv) * stride) * 512 + col;
    o[g] = (h16)oacc[vv];
  }
}

// ---------------------------------------------------------------------------
// Host orchestration
// ---------------------------------------------------------------------------
extern "C" void kernel_launch(void* const* d_in, const int* in_sizes, int n_in,
                              void* d_out, int out_size, void* d_ws, size_t ws_size,
                              hipStream_t stream) {
  (void)in_sizes; (void)n_in; (void)out_size; (void)ws_size;
  const float* x   = (const float*)d_in[0];
  const float* lns = (const float*)d_in[1];
  const float* lnb = (const float*)d_in[2];
  const float* wq  = (const float*)d_in[3];
  const float* bq  = (const float*)d_in[4];
  const float* wk  = (const float*)d_in[5];
  const float* bk  = (const float*)d_in[6];
  const float* wv  = (const float*)d_in[7];
  const float* bv  = (const float*)d_in[8];
  const float* wo  = (const float*)d_in[9];
  const float* bo  = (const float*)d_in[10];
  const float* lms = (const float*)d_in[11];
  const float* lmb = (const float*)d_in[12];
  const float* w1  = (const float*)d_in[13];
  const float* b1  = (const float*)d_in[14];
  const float* w2  = (const float*)d_in[15];
  const float* b2  = (const float*)d_in[16];

  const size_t NTOK = 32768, C = 512, MLPD = 2048;
  float* resid = (float*)d_out;

  char* ws = (char*)d_ws;
  h16* xln  = (h16*)ws; ws += NTOK * C * 2;
  h16* qbuf = (h16*)ws; ws += NTOK * C * 2;
  h16* kbuf = (h16*)ws; ws += NTOK * C * 2;
  h16* vbuf = (h16*)ws; ws += NTOK * C * 2;
  h16* abuf = (h16*)ws; ws += NTOK * C * 2;
  h16* hid  = (h16*)ws; ws += NTOK * MLPD * 2;
  h16* wqT  = (h16*)ws; ws += 3 * C * C * 2;
  h16* wkT  = (h16*)ws; ws += 3 * C * C * 2;
  h16* wvT  = (h16*)ws; ws += 3 * C * C * 2;
  h16* woT  = (h16*)ws; ws += 3 * C * C * 2;
  h16* w1T  = (h16*)ws; ws += C * MLPD * 2;
  h16* w2T  = (h16*)ws; ws += C * MLPD * 2;

  // residual stream starts as x (fp32, in place in d_out)
  hipMemcpyAsync(resid, x, NTOK * C * sizeof(float), hipMemcpyDeviceToDevice, stream);

  // weight prep: transpose + convert to f16 [N,K]
  {
    const int tn = (int)(C * C);
    dim3 tg((tn + 255) / 256);
    for (int i = 0; i < 3; ++i) {
      tconv_kernel<<<tg, 256, 0, stream>>>(wq + (size_t)i * tn, wqT + (size_t)i * tn, C, C);
      tconv_kernel<<<tg, 256, 0, stream>>>(wk + (size_t)i * tn, wkT + (size_t)i * tn, C, C);
      tconv_kernel<<<tg, 256, 0, stream>>>(wv + (size_t)i * tn, wvT + (size_t)i * tn, C, C);
      tconv_kernel<<<tg, 256, 0, stream>>>(wo + (size_t)i * tn, woT + (size_t)i * tn, C, C);
    }
    tconv_kernel<<<((int)(C * MLPD) + 255) / 256, 256, 0, stream>>>(w1, w1T, C, MLPD);
    tconv_kernel<<<((int)(C * MLPD) + 255) / 256, 256, 0, stream>>>(w2, w2T, MLPD, C);
  }

  // axis params: S, stride, shift, mul, nseq
  const int axS[3]      = {16,   32,   32};
  const int axStride[3] = {1024, 32,   1};
  const int axShift[3]  = {10,   5,    0};
  const int axMul[3]    = {16384, 1024, 32};
  const int axNseq[3]   = {2048, 1024, 1024};

  const dim3 ggC(C / 64, NTOK / 128);   // 8 x 256 blocks, 256 thr each

  for (int i = 0; i < 3; ++i) {
    ln_to_f16<<<NTOK / 8, 256, 0, stream>>>(resid, lns + (size_t)i * C,
                                            lnb + (size_t)i * C, xln, NTOK);
    gemm_wmma_tiled<0><<<ggC, 256, 0, stream>>>(xln, wqT + (size_t)i * C * C,
                                                bq + (size_t)i * C, qbuf, nullptr,
                                                NTOK, C, C, 0.125f);
    gemm_wmma_tiled<0><<<ggC, 256, 0, stream>>>(xln, wkT + (size_t)i * C * C,
                                                bk + (size_t)i * C, kbuf, nullptr,
                                                NTOK, C, C, 1.0f);
    gemm_wmma_tiled<0><<<ggC, 256, 0, stream>>>(xln, wvT + (size_t)i * C * C,
                                                bv + (size_t)i * C, vbuf, nullptr,
                                                NTOK, C, C, 1.0f);
    if (axS[i] == 16)
      axial_attn<16><<<dim3(8, axNseq[i]), 128, 0, stream>>>(qbuf, kbuf, vbuf, abuf,
                                                             axStride[i], axShift[i], axMul[i]);
    else
      axial_attn<32><<<dim3(8, axNseq[i]), 256, 0, stream>>>(qbuf, kbuf, vbuf, abuf,
                                                             axStride[i], axShift[i], axMul[i]);
    gemm_wmma_tiled<1><<<ggC, 256, 0, stream>>>(abuf, woT + (size_t)i * C * C,
                                                bo + (size_t)i * C, nullptr, resid,
                                                NTOK, C, C, 1.0f);
  }

  // final LN + MLP (gelu up-proj f16, down-proj accumulates into residual)
  ln_to_f16<<<NTOK / 8, 256, 0, stream>>>(resid, lms, lmb, xln, NTOK);
  gemm_wmma_tiled<2><<<dim3(MLPD / 64, NTOK / 128), 256, 0, stream>>>(
      xln, w1T, b1, hid, nullptr, NTOK, MLPD, C, 1.0f);
  gemm_wmma_tiled<1><<<dim3(C / 64, NTOK / 128), 256, 0, stream>>>(
      hid, w2T, b2, nullptr, resid, NTOK, C, MLPD, 1.0f);
}
